// fGraphAttentionHead_3135326126436
// MI455X (gfx1250) — matrix operand order repaired and checked by reading, not compiled, not run on hardware
//
#include <hip/hip_runtime.h>
#include <hip/hip_bf16.h>
#include <math.h>

#define IN_F   256
#define OUT_F  32
#define ALPHA  0.2f

typedef __attribute__((ext_vector_type(2)))  float    v2f;
typedef __attribute__((ext_vector_type(8)))  float    v8f;
typedef __attribute__((ext_vector_type(16))) _Float16 v16h;

// ---------------------------------------------------------------------------
// Kernel 1: v_src = fW @ a_src, v_dest = fW @ a_dest   (256x32 @ 32x1)
// ---------------------------------------------------------------------------
__global__ void k_vecs(const float* __restrict__ fW,
                       const float* __restrict__ a_src,
                       const float* __restrict__ a_dest,
                       float* __restrict__ v_src,
                       float* __restrict__ v_dest) {
    int i = threadIdx.x;            // 0..255 (one row of fW each)
    float s = 0.f, d = 0.f;
#pragma unroll
    for (int j = 0; j < OUT_F; ++j) {
        float w = fW[i * OUT_F + j];
        s += w * a_src[j];
        d += w * a_dest[j];
    }
    v_src[i]  = s;
    v_dest[i] = d;
}

// ---------------------------------------------------------------------------
// Kernel 2: f1[i] = dot(from_feat[i,:], v_src); f2[i] = dot(to_feat[i,:], v_dest)
// One wave32 per node; coalesced 128B loads; shfl reduction.
// ---------------------------------------------------------------------------
__global__ void k_gemv(const float* __restrict__ from_feat,
                       const float* __restrict__ to_feat,
                       const float* __restrict__ v_src,
                       const float* __restrict__ v_dest,
                       float* __restrict__ f1,
                       float* __restrict__ f2,
                       int N) {
    int wave = (int)((blockIdx.x * blockDim.x + threadIdx.x) >> 5);
    int lane = threadIdx.x & 31;
    if (wave >= N) return;
    const float* fr = from_feat + (size_t)wave * IN_F;
    const float* to = to_feat   + (size_t)wave * IN_F;
    float s1 = 0.f, s2 = 0.f;
#pragma unroll
    for (int j = 0; j < IN_F / 32; ++j) {
        int idx = lane + j * 32;
        s1 += fr[idx] * v_src[idx];
        s2 += to[idx] * v_dest[idx];
    }
#pragma unroll
    for (int off = 16; off > 0; off >>= 1) {
        s1 += __shfl_xor(s1, off, 32);
        s2 += __shfl_xor(s2, off, 32);
    }
    if (lane == 0) { f1[wave] = s1; f2[wave] = s2; }
}

// ---------------------------------------------------------------------------
// Kernel 3: Wh = h @ W   [N x 256] @ [256 x 32], WMMA.
// Per wave: 16 rows x 32 cols (two 16x16 accumulators). 8 waves -> 128 rows/block.
// B is staged in LDS pre-swizzled as K-pairs so every B fragment is one
// contiguous 8B ds_load_b64 into an even-aligned VGPR pair (no repack moves).
// ---------------------------------------------------------------------------
#if __has_builtin(__builtin_amdgcn_wmma_f32_16x16x4_f32)
// ---- full-precision path: V_WMMA_F32_16X16X4_F32 ----
__global__ void __launch_bounds__(256)
k_wh(const float* __restrict__ h, const float* __restrict__ W,
     float* __restrict__ Wh, int N) {
    // Wsw[kp*32 + n] = { W[2*kp][n], W[2*kp+1][n] }   (kp = 0..127)
    __shared__ v2f Wsw[(IN_F / 2) * OUT_F];       // 32 KB of 320 KB WGP LDS
    for (int t = threadIdx.x; t < (IN_F / 2) * OUT_F; t += blockDim.x) {
        int kp = t >> 5;            // K pair index
        int c  = t & 31;            // column
        v2f w;
        w.x = W[(2 * kp)     * OUT_F + c];
        w.y = W[(2 * kp + 1) * OUT_F + c];
        Wsw[t] = w;
    }
    __syncthreads();

    int waveId  = threadIdx.x >> 5;               // 0..7
    int lane    = threadIdx.x & 31;
    int rowBase = blockIdx.x * 128 + waveId * 16;
    int mrow    = lane & 15;
    int khalf   = (lane >> 4) * 2;                // lanes 0-15: K+0/K+1, 16-31: K+2/K+3
    int ncol    = lane & 15;

    int arow = rowBase + mrow;
    if (arow >= N) arow = N - 1;                  // clamp (keeps EXEC all-ones)
    const float* hrow = h + (size_t)arow * IN_F;

    v8f c0 = {}; v8f c1 = {};
#pragma unroll 8
    for (int k = 0; k < IN_F; k += 4) {
        int kb = k + khalf;                       // always even
        v2f a  = *(const v2f*)(hrow + kb);        // aligned 8B global load
        v2f b0 = Wsw[(kb >> 1) * OUT_F + ncol];        // one ds_load_b64
        v2f b1 = Wsw[(kb >> 1) * OUT_F + 16 + ncol];   // one ds_load_b64
        c0 = __builtin_amdgcn_wmma_f32_16x16x4_f32(false, a, false, b0, (short)0, c0, false, false);
        c1 = __builtin_amdgcn_wmma_f32_16x16x4_f32(false, a, false, b1, (short)0, c1, false, false);
    }

    int half = lane >> 4;                         // C layout: VGPR j -> rows j / j+8
#pragma unroll
    for (int j = 0; j < 8; ++j) {
        int r = rowBase + j + half * 8;
        if (r < N) {
            Wh[(size_t)r * OUT_F + ncol]      = c0[j];
            Wh[(size_t)r * OUT_F + 16 + ncol] = c1[j];
        }
    }
}
#else
// ---- fallback: codegen-confirmed V_WMMA_F32_16X16X32_F16 (memory-bound, f16 ok) ----
__global__ void __launch_bounds__(256)
k_wh(const float* __restrict__ h, const float* __restrict__ W,
     float* __restrict__ Wh, int N) {
    __shared__ _Float16 Ws[IN_F * OUT_F];         // 16 KB
    for (int t = threadIdx.x; t < IN_F * OUT_F; t += blockDim.x)
        Ws[t] = (_Float16)W[t];
    __syncthreads();

    int waveId  = threadIdx.x >> 5;
    int lane    = threadIdx.x & 31;
    int rowBase = blockIdx.x * 128 + waveId * 16;
    int mrow    = lane & 15;
    int laneHi  = lane >> 4;
    int ncol    = lane & 15;

    int arow = rowBase + mrow;
    if (arow >= N) arow = N - 1;
    const float* hrow = h + (size_t)arow * IN_F;

    v8f c0 = {}; v8f c1 = {};
    for (int k = 0; k < IN_F; k += 32) {
        v16h a, b0, b1;
#pragma unroll
        for (int t = 0; t < 16; ++t) {
            int v = t >> 1, p = t & 1;
            int kk = t + laneHi * 16; (void)v; (void)p;
            a[t] = (_Float16)hrow[k + kk];
            int kb = k + t + 16 * laneHi;         // B: lanes 0-15 K=0..15, 16-31 K=16..31
            b0[t] = Ws[kb * OUT_F + ncol];
            b1[t] = Ws[kb * OUT_F + 16 + ncol];
        }
        c0 = __builtin_amdgcn_wmma_f32_16x16x32_f16(false, a, false, b0, (short)0, c0, false, false);
        c1 = __builtin_amdgcn_wmma_f32_16x16x32_f16(false, a, false, b1, (short)0, c1, false, false);
    }
#pragma unroll
    for (int j = 0; j < 8; ++j) {
        int r = rowBase + j + laneHi * 8;
        if (r < N) {
            Wh[(size_t)r * OUT_F + ncol]      = c0[j];
            Wh[(size_t)r * OUT_F + 16 + ncol] = c1[j];
        }
    }
}
#endif

// ---------------------------------------------------------------------------
// Kernel 4: fused per-node attention softmax + SpMM + ELU.
// One wave per node; `row` is sorted -> binary search the edge range.
// Pass 1 (edge-parallel across lanes): segment max.
// Pass 2 (sequential edges, lane = output feature): sum exp & accumulate Wh rows.
// ---------------------------------------------------------------------------
__global__ void k_node(const int* __restrict__ row, const int* __restrict__ col,
                       const float* __restrict__ f1, const float* __restrict__ f2,
                       const float* __restrict__ Wh, float* __restrict__ out,
                       int N, int E) {
    int node = (int)((blockIdx.x * blockDim.x + threadIdx.x) >> 5);
    int lane = threadIdx.x & 31;
    if (node >= N) return;

    // lower_bound(row, node) and lower_bound(row, node+1) — row[] is L2-hot.
    int lo = 0, hi = E;
    while (lo < hi) { int mid = (lo + hi) >> 1; if (row[mid] < node) lo = mid + 1; else hi = mid; }
    int start = lo;
    hi = E;
    while (lo < hi) { int mid = (lo + hi) >> 1; if (row[mid] < node + 1) lo = mid + 1; else hi = mid; }
    int end = lo;

    if (end == start) { out[(size_t)node * OUT_F + lane] = 0.f; return; }

    float f1r = f1[node];

    // pass 1: segment max over leaky_relu(f1[r] + f2[c])
    float m = -3.402823466e38f;
    for (int e = start + lane; e < end; e += 32) {
        float v = f1r + f2[col[e]];
        v = v > 0.f ? v : ALPHA * v;
        m = fmaxf(m, v);
    }
#pragma unroll
    for (int off = 16; off > 0; off >>= 1) m = fmaxf(m, __shfl_xor(m, off, 32));

    // pass 2: sum of exp + weighted accumulation of Wh rows (lane = feature)
    float sum = 0.f, hp = 0.f;
    for (int e = start; e < end; ++e) {
        int c = col[e];                            // uniform across the wave
        float v = f1r + f2[c];
        v = v > 0.f ? v : ALPHA * v;
        float w = expf(v - m);
        sum += w;
        hp += w * Wh[(size_t)c * OUT_F + lane];    // coalesced 128B row, L2-resident
    }
    float x = hp / sum;
    out[(size_t)node * OUT_F + lane] = x > 0.f ? x : (expf(x) - 1.f);
}

// ---------------------------------------------------------------------------
extern "C" void kernel_launch(void* const* d_in, const int* in_sizes, int n_in,
                              void* d_out, int out_size, void* d_ws, size_t ws_size,
                              hipStream_t stream) {
    const float* h         = (const float*)d_in[0];
    const float* from_feat = (const float*)d_in[1];
    const float* to_feat   = (const float*)d_in[2];
    const int*   row       = (const int*)  d_in[3];
    const int*   col       = (const int*)  d_in[4];
    const float* W         = (const float*)d_in[5];
    const float* fW        = (const float*)d_in[6];
    const float* a_src     = (const float*)d_in[7];
    const float* a_dest    = (const float*)d_in[8];
    float*       out       = (float*)d_out;

    const int N = in_sizes[0] / IN_F;
    const int E = in_sizes[3];

    // workspace layout (floats)
    float* ws     = (float*)d_ws;
    float* v_src  = ws;                    // 256
    float* v_dest = ws + 256;              // 256
    float* f1     = ws + 512;              // N
    float* f2     = f1 + N;                // N
    float* Wh     = f2 + N;                // N * 32

    k_vecs<<<1, 256, 0, stream>>>(fW, a_src, a_dest, v_src, v_dest);

    {
        long long threads = (long long)N * 32;
        int blocks = (int)((threads + 255) / 256);
        k_gemv<<<blocks, 256, 0, stream>>>(from_feat, to_feat, v_src, v_dest, f1, f2, N);
    }

    {
        int blocks = (N + 127) / 128;      // 128 rows per block (8 waves x 16 rows)
        k_wh<<<blocks, 256, 0, stream>>>(h, W, Wh, N);
    }

    {
        long long threads = (long long)N * 32;
        int blocks = (int)((threads + 255) / 256);
        k_node<<<blocks, 256, 0, stream>>>(row, col, f1, f2, Wh, out, N, E);
    }
}
